// Attention_22299470201540
// MI455X (gfx1250) — matrix operand order repaired and checked by reading, not compiled, not run on hardware
//
#include <hip/hip_runtime.h>
#include <hip/hip_bf16.h>

typedef __attribute__((ext_vector_type(16))) _Float16 v16h;
typedef __attribute__((ext_vector_type(8)))  float    v8f;
typedef _Float16 half_t;

#define BQ 2
#define NQ 2048
#define DIMQ 1536
#define HQ 8
#define DKQ 64
#define DVQ 192
#define HDV 1536      // H*DV
#define QKVCOLS 2560  // 512 + 512 + 1536

// ---------------------------------------------------------------------------
// CDNA5 async global->LDS path (ASYNCcnt), guarded so either toolchain builds.
// Builtin signature (from hipcc diagnostic): 
//   (v4i __device__*, v4i __shared__*, int offset, int cpol)
// ---------------------------------------------------------------------------
#if defined(__has_builtin)
#if __has_builtin(__builtin_amdgcn_global_load_async_to_lds_b128) && \
    __has_builtin(__builtin_amdgcn_s_wait_asynccnt)
#define USE_ASYNC_COPY 1
#endif
#endif
#ifndef USE_ASYNC_COPY
#define USE_ASYNC_COPY 0
#endif

#if USE_ASYNC_COPY
typedef int async_v4i __attribute__((vector_size(16)));
#define ASYNC_WAIT(n) __builtin_amdgcn_s_wait_asynccnt(n)
#else
#define ASYNC_WAIT(n)
#endif

// Copy a 4KB K-tile (32 rows x 64 f16) global -> LDS; 16B per lane per issue.
__device__ static inline void copy_ktile(const half_t* __restrict__ g,
                                         half_t* __restrict__ l) {
  const int lane = threadIdx.x & 31;
#if USE_ASYNC_COPY
#pragma unroll
  for (int i = 0; i < 8; ++i) {
    const int e = (i * 32 + lane) * 8;  // 8 halfs = 16 bytes
    __builtin_amdgcn_global_load_async_to_lds_b128(
        (__attribute__((address_space(1))) async_v4i*)(g + e),
        (__attribute__((address_space(3))) async_v4i*)(l + e), 0, 0);
  }
#else
#pragma unroll
  for (int i = 0; i < 8; ++i) {
    const int e = (i * 32 + lane) * 8;
    *reinterpret_cast<int4*>(l + e) = *reinterpret_cast<const int4*>(g + e);
  }
#endif
}

// ---------------------------------------------------------------------------
// WMMA helpers (CDNA5 wave32, 16x16x32 f16 -> f32)
// ---------------------------------------------------------------------------
__device__ static inline v8f wmma_f16(v16h a, v16h b, v8f c) {
  return __builtin_amdgcn_wmma_f32_16x16x32_f16(false, a, false, b, (short)0, c,
                                                false, false);
}

// A fragment (16x32, f16): lane<16 holds row m=lane, K = {0..7,16..23};
// lane>=16 holds row m=lane-16, K = {8..15,24..31}.  (ISA 7.12.2)
__device__ static inline v16h afrag_f16(const half_t* base, int ld, int k0) {
  const int l  = threadIdx.x & 31;
  const int m  = l & 15;
  const int ks = (l < 16) ? 0 : 8;
  const half_t* p = base + (size_t)m * ld + k0 + ks;
  v16h a;
  *reinterpret_cast<int4*>(&a)     = *reinterpret_cast<const int4*>(p);
  *(reinterpret_cast<int4*>(&a)+1) = *reinterpret_cast<const int4*>(p + 16);
  return a;
}

// B fragment (32x16, f16) from a row-major transposed matrix bt[n][k]:
// lane<16: col n = n0+lane, K = k0+{0..7,16..23} ; lane>=16: col n0+lane-16, K = k0+{8..15,24..31}.
__device__ static inline v16h bfragT(const half_t* bt, int ld, int n0, int k0) {
  const int l  = threadIdx.x & 31;
  const int n  = n0 + (l & 15);
  const int ko = k0 + ((l < 16) ? 0 : 8);
  const half_t* p = bt + (size_t)n * ld + ko;
  v16h b;
  *reinterpret_cast<int4*>(&b)     = *reinterpret_cast<const int4*>(p);
  *(reinterpret_cast<int4*>(&b)+1) = *reinterpret_cast<const int4*>(p + 16);
  return b;
}

// ---------------------------------------------------------------------------
// Kernel 0: f32->f16 conversions: transposed weights + x.
// wt_qkv[j][k]: j<512 -> Wq col j ; j<1024 -> Wk col j-512 ; else Wv col j-1024.
// wot[j][k] = Wo[k][j].  xh = f16(x) (same layout).
// ---------------------------------------------------------------------------
__global__ void prep_inputs(const float* __restrict__ Wq,
                            const float* __restrict__ Wk,
                            const float* __restrict__ Wv,
                            const float* __restrict__ Wo,
                            const float* __restrict__ x,
                            half_t* __restrict__ wt_qkv,
                            half_t* __restrict__ wot,
                            half_t* __restrict__ xh) {
  size_t idx = (size_t)blockIdx.x * blockDim.x + threadIdx.x;
  const size_t nqkv = (size_t)QKVCOLS * DIMQ;
  const size_t no   = (size_t)HDV * DIMQ;
  const size_t nx   = (size_t)BQ * NQ * DIMQ;
  if (idx < nqkv) {
    int j = (int)(idx / DIMQ);
    int k = (int)(idx % DIMQ);
    float v;
    if (j < 512)       v = Wq[(size_t)k * 512 + j];
    else if (j < 1024) v = Wk[(size_t)k * 512 + (j - 512)];
    else               v = Wv[(size_t)k * HDV + (j - 1024)];
    wt_qkv[idx] = (half_t)v;
  } else if (idx < nqkv + no) {
    size_t t = idx - nqkv;
    int j = (int)(t / DIMQ);
    int k = (int)(t % DIMQ);
    wot[t] = (half_t)Wo[(size_t)k * DIMQ + j];
  } else if (idx < nqkv + no + nx) {
    size_t t = idx - nqkv - no;
    xh[t] = (half_t)x[t];
  }
}

// ---------------------------------------------------------------------------
// Kernel 1: fused QKV projection GEMM. One wave computes a 64x64 tile of
// xh[4096,1536] @ W[1536,2560].  Q gets scale + content bias folded in;
// V is stored transposed [b,h,dv,n] for contiguous P@V B-fragments.
// ---------------------------------------------------------------------------
__global__ __launch_bounds__(32) void qkv_gemm(const half_t* __restrict__ xh,
                                               const half_t* __restrict__ wt,
                                               const float* __restrict__ cbias,
                                               half_t* __restrict__ Qh,
                                               half_t* __restrict__ Kh,
                                               half_t* __restrict__ Vt) {
  const int row0 = blockIdx.x * 64;
  const int col0 = blockIdx.y * 64;
  v8f c[4][4] = {};
  for (int k0 = 0; k0 < DIMQ; k0 += 32) {
    v16h b[4];
#pragma unroll
    for (int nt = 0; nt < 4; ++nt)
      b[nt] = bfragT(wt, DIMQ, col0 + nt * 16, k0);
#pragma unroll
    for (int mt = 0; mt < 4; ++mt) {
      v16h a = afrag_f16(xh + (size_t)(row0 + mt * 16) * DIMQ, DIMQ, k0);
#pragma unroll
      for (int nt = 0; nt < 4; ++nt)
        c[mt][nt] = wmma_f16(a, b[nt], c[mt][nt]);
    }
  }
  const int l     = threadIdx.x & 31;
  const int ncol  = l & 15;
  const int mbase = (l < 16) ? 0 : 8;
#pragma unroll
  for (int mt = 0; mt < 4; ++mt) {
#pragma unroll
    for (int nt = 0; nt < 4; ++nt) {
      const int j = col0 + nt * 16 + ncol;
#pragma unroll
      for (int r = 0; r < 8; ++r) {
        const int i  = row0 + mt * 16 + mbase + r;
        const int bb = i >> 11;      // / 2048
        const int n  = i & 2047;
        float v = c[mt][nt][r];
        if (j < 512) {
          const int h = j >> 6, dk = j & 63;
          v = v * 0.125f + cbias[(h << 6) + dk];   // DK^-0.5 = 0.125
          Qh[(((size_t)bb * HQ + h) * NQ + n) * DKQ + dk] = (half_t)v;
        } else if (j < 1024) {
          const int jj = j - 512;
          const int h = jj >> 6, dk = jj & 63;
          Kh[(((size_t)bb * HQ + h) * NQ + n) * DKQ + dk] = (half_t)v;
        } else {
          const int jj = j - 1024;
          const int h = jj / DVQ, dv = jj % DVQ;
          Vt[(((size_t)bb * HQ + h) * DVQ + dv) * NQ + n] = (half_t)v;
        }
      }
    }
  }
}

// ---------------------------------------------------------------------------
// Kernel 2: per-row rank-1 rel-pos coefficient
//   c[b,h,n] = sum_d (Qh - content_bias + pos_bias) * W_rel
// (Qh already holds q*scale + content_bias, so this equals (q*scale+pb)·Wrel.)
// ---------------------------------------------------------------------------
__global__ void compute_c(const half_t* __restrict__ Qh,
                          const float* __restrict__ Wrel,
                          const float* __restrict__ cbias,
                          const float* __restrict__ pbias,
                          float* __restrict__ cArr) {
  const int idx = blockIdx.x * blockDim.x + threadIdx.x;
  if (idx >= BQ * HQ * NQ) return;
  const int h = (idx >> 11) & (HQ - 1);
  const half_t* q = Qh + (size_t)idx * DKQ;
  const float* wr = Wrel + h * DKQ;
  const float* pb = pbias + h * DKQ;
  const float* cb = cbias + h * DKQ;
  float s = 0.f;
#pragma unroll 8
  for (int d = 0; d < DKQ; ++d)
    s += ((float)q[d] + pb[d] - cb[d]) * wr[d];
  cArr[idx] = s;
}

// ---------------------------------------------------------------------------
// Kernel 3: flash attention with analytic relative-shift term.
// One wave owns a 16-row query tile; streams 32 keys/values per step.
//   logits[i,j] = Qc_i . K_j + (j-i)*c_i ;  online softmax ;  O += P V.
// K tiles are double-buffered global->LDS via async copies (ASYNCcnt);
// C-fragment -> A-fragment relayout for P goes through LDS.
// ---------------------------------------------------------------------------
__global__ __launch_bounds__(32) void attn_kernel(const half_t* __restrict__ Qh,
                                                  const half_t* __restrict__ Kh,
                                                  const half_t* __restrict__ Vt,
                                                  const float* __restrict__ cArr,
                                                  half_t* __restrict__ Oh) {
  __shared__ __align__(16) half_t ldsK[2][32 * DKQ];  // 2 x 4KB K tiles
  __shared__ __align__(16) half_t ldsP[16 * 32];
  const int it = blockIdx.x;          // query tile (0..127)
  const int bh = blockIdx.y;          // 0..15
  const int b  = bh >> 3, h = bh & 7;
  const int i0 = it * 16;
  const int l     = threadIdx.x & 31;
  const int ncol  = l & 15;
  const int mbase = (l < 16) ? 0 : 8;

  const half_t* qbase = Qh + (size_t)bh * NQ * DKQ;
  const half_t* kbase = Kh + (size_t)bh * NQ * DKQ;
  const half_t* vbase = Vt + (size_t)bh * DVQ * NQ;
  const float*  cbp   = cArr + (size_t)bh * NQ + i0;

  // Q A-fragments for the whole tile (DK = 64 -> two K-chunks)
  const v16h aq0 = afrag_f16(qbase + (size_t)i0 * DKQ, DKQ, 0);
  const v16h aq1 = afrag_f16(qbase + (size_t)i0 * DKQ, DKQ, 32);

  float cv[8], m_run[8], l_run[8];
#pragma unroll
  for (int r = 0; r < 8; ++r) {
    cv[r]    = cbp[mbase + r];
    m_run[r] = -3.0e38f;
    l_run[r] = 0.f;
  }
  v8f o[12] = {};

  // stage first K tile
  copy_ktile(kbase, &ldsK[0][0]);
  int cur = 0;

  for (int j0 = 0; j0 < NQ; j0 += 32) {
    if (j0 + 32 < NQ) {
      copy_ktile(kbase + (size_t)(j0 + 32) * DKQ, &ldsK[cur ^ 1][0]);
      ASYNC_WAIT(8);   // current tile complete; next tile may stay in flight
      __builtin_prefetch(vbase + (j0 + 32), 0, 1);  // global_prefetch (V stream)
    } else {
      ASYNC_WAIT(0);
    }

    // ---- S = Qc @ K^T  (two 16x16 column tiles, K from LDS) ----
    v8f s[2];
#pragma unroll
    for (int jt = 0; jt < 2; ++jt) {
      v8f acc = {};
      acc = wmma_f16(aq0, bfragT(&ldsK[cur][0], DKQ, jt * 16, 0),  acc);
      acc = wmma_f16(aq1, bfragT(&ldsK[cur][0], DKQ, jt * 16, 32), acc);
      s[jt] = acc;
    }
    cur ^= 1;

    // ---- rel term + row max ----
    float rm[8];
#pragma unroll
    for (int r = 0; r < 8; ++r) {
      const float irow = (float)(i0 + mbase + r);
      const float s0 = s[0][r] + ((float)(j0 + ncol)      - irow) * cv[r];
      const float s1 = s[1][r] + ((float)(j0 + 16 + ncol) - irow) * cv[r];
      s[0][r] = s0; s[1][r] = s1;
      rm[r] = fmaxf(s0, s1);
    }
#pragma unroll
    for (int off = 8; off >= 1; off >>= 1)
#pragma unroll
      for (int r = 0; r < 8; ++r)
        rm[r] = fmaxf(rm[r], __shfl_xor(rm[r], off, 32));
    // ---- online softmax update ----
    float sc[8], rs[8];
#pragma unroll
    for (int r = 0; r < 8; ++r) {
      const float mn = fmaxf(m_run[r], rm[r]);
      sc[r] = __expf(m_run[r] - mn);
      m_run[r] = mn;
      const float p0 = __expf(s[0][r] - mn);
      const float p1 = __expf(s[1][r] - mn);
      s[0][r] = p0; s[1][r] = p1;
      rs[r] = p0 + p1;
    }
#pragma unroll
    for (int off = 8; off >= 1; off >>= 1)
#pragma unroll
      for (int r = 0; r < 8; ++r)
        rs[r] += __shfl_xor(rs[r], off, 32);
#pragma unroll
    for (int r = 0; r < 8; ++r) l_run[r] = l_run[r] * sc[r] + rs[r];
#pragma unroll
    for (int t = 0; t < 12; ++t)
#pragma unroll
      for (int r = 0; r < 8; ++r) o[t][r] *= sc[r];

    // ---- P: C-fragment layout -> LDS -> A-fragment layout ----
    __syncthreads();
#pragma unroll
    for (int jt = 0; jt < 2; ++jt)
#pragma unroll
      for (int r = 0; r < 8; ++r)
        ldsP[(mbase + r) * 32 + jt * 16 + ncol] = (half_t)s[jt][r];
    __syncthreads();
    v16h ap;
    {
      const int ks = (l < 16) ? 0 : 8;
      const half_t* p = &ldsP[ncol * 32 + ks];
      *reinterpret_cast<int4*>(&ap)     = *reinterpret_cast<const int4*>(p);
      *(reinterpret_cast<int4*>(&ap)+1) = *reinterpret_cast<const int4*>(p + 16);
    }
    // ---- O += P @ V  (DV = 192 -> 12 column tiles) ----
#pragma unroll
    for (int t = 0; t < 12; ++t) {
      const v16h bv = bfragT(vbase, NQ, t * 16, j0);
      o[t] = wmma_f16(ap, bv, o[t]);
    }
  }

  // ---- normalize + store f16 [b, n, h*DV + dv] ----
#pragma unroll
  for (int r = 0; r < 8; ++r) l_run[r] = 1.f / l_run[r];
#pragma unroll
  for (int t = 0; t < 12; ++t) {
#pragma unroll
    for (int r = 0; r < 8; ++r) {
      const int i   = i0 + mbase + r;
      const int col = h * DVQ + t * 16 + ncol;
      Oh[((size_t)b * NQ + i) * HDV + col] = (half_t)(o[t][r] * l_run[r]);
    }
  }
}

// ---------------------------------------------------------------------------
// Kernel 4: output projection  out = Oh[4096,1536] @ Wo + bo  (f32 out)
// ---------------------------------------------------------------------------
__global__ __launch_bounds__(32) void out_gemm(const half_t* __restrict__ Oh,
                                               const half_t* __restrict__ wot,
                                               const float* __restrict__ bo,
                                               float* __restrict__ out) {
  const int row0 = blockIdx.x * 64;
  const int col0 = blockIdx.y * 64;
  v8f c[4][4] = {};
  for (int k0 = 0; k0 < HDV; k0 += 32) {
    v16h b[4];
#pragma unroll
    for (int nt = 0; nt < 4; ++nt)
      b[nt] = bfragT(wot, HDV, col0 + nt * 16, k0);
#pragma unroll
    for (int mt = 0; mt < 4; ++mt) {
      v16h a = afrag_f16(Oh + (size_t)(row0 + mt * 16) * HDV, HDV, k0);
#pragma unroll
      for (int nt = 0; nt < 4; ++nt)
        c[mt][nt] = wmma_f16(a, b[nt], c[mt][nt]);
    }
  }
  const int l     = threadIdx.x & 31;
  const int ncol  = l & 15;
  const int mbase = (l < 16) ? 0 : 8;
#pragma unroll
  for (int mt = 0; mt < 4; ++mt)
#pragma unroll
    for (int nt = 0; nt < 4; ++nt) {
      const int j = col0 + nt * 16 + ncol;
#pragma unroll
      for (int r = 0; r < 8; ++r) {
        const int i = row0 + mt * 16 + mbase + r;
        out[(size_t)i * DIMQ + j] = c[mt][nt][r] + bo[j];
      }
    }
}

// ---------------------------------------------------------------------------
extern "C" void kernel_launch(void* const* d_in, const int* in_sizes, int n_in,
                              void* d_out, int out_size, void* d_ws, size_t ws_size,
                              hipStream_t stream) {
  const float* x     = (const float*)d_in[0];
  const float* Wq    = (const float*)d_in[1];
  const float* Wk    = (const float*)d_in[2];
  const float* Wv    = (const float*)d_in[3];
  const float* Wrel  = (const float*)d_in[4];
  const float* Wo    = (const float*)d_in[5];
  const float* bo    = (const float*)d_in[6];
  const float* cbias = (const float*)d_in[7];   // rel_content_bias [H*DK]
  const float* pbias = (const float*)d_in[8];   // rel_pos_bias     [H*DK]
  float* out = (float*)d_out;

  // workspace carve-up (all offsets 256B aligned)
  char* ws = (char*)d_ws;
  half_t* wt_qkv = (half_t*)ws; ws += (size_t)QKVCOLS * DIMQ * 2;      // 7.9 MB
  half_t* wot    = (half_t*)ws; ws += (size_t)HDV * DIMQ * 2;         // 4.7 MB
  half_t* xh     = (half_t*)ws; ws += (size_t)BQ * NQ * DIMQ * 2;     // 12.6 MB
  half_t* Qh     = (half_t*)ws; ws += (size_t)BQ * HQ * NQ * DKQ * 2; // 4.2 MB
  half_t* Kh     = (half_t*)ws; ws += (size_t)BQ * HQ * NQ * DKQ * 2; // 4.2 MB
  half_t* Vt     = (half_t*)ws; ws += (size_t)BQ * HQ * DVQ * NQ * 2; // 12.6 MB
  float*  cArr   = (float*) ws; ws += (size_t)BQ * HQ * NQ * 4;       // 131 KB
  half_t* Oh     = (half_t*)ws; ws += (size_t)BQ * NQ * HDV * 2;      // 12.6 MB

  // 0: weight transpose + f16 conversion of weights and x
  {
    const size_t total = (size_t)QKVCOLS * DIMQ + (size_t)HDV * DIMQ +
                         (size_t)BQ * NQ * DIMQ;
    const int blocks = (int)((total + 255) / 256);
    prep_inputs<<<blocks, 256, 0, stream>>>(Wq, Wk, Wv, Wo, x, wt_qkv, wot, xh);
  }
  // 1: fused QKV projection (WMMA)
  qkv_gemm<<<dim3(BQ * NQ / 64, QKVCOLS / 64), 32, 0, stream>>>(
      xh, wt_qkv, cbias, Qh, Kh, Vt);
  // 2: rank-1 rel-pos coefficients
  compute_c<<<(BQ * HQ * NQ + 255) / 256, 256, 0, stream>>>(
      Qh, Wrel, cbias, pbias, cArr);
  // 3: flash attention (WMMA + async LDS K tiles)
  attn_kernel<<<dim3(NQ / 16, BQ * HQ), 32, 0, stream>>>(Qh, Kh, Vt, cArr, Oh);
  // 4: output projection (WMMA)
  out_gemm<<<dim3(BQ * NQ / 64, DIMQ / 64), 32, 0, stream>>>(Oh, wot, bo, out);
}
